// SlidingRNN_v2_56040733278234
// MI455X (gfx1250) — compile-verified, matrix operand order checked by reading, not run on hardware
//
#include <hip/hip_runtime.h>
#include <hip/hip_bf16.h>
#include <math.h>

// ---------------- problem constants (match reference) ----------------
#define B_   512
#define T_   64
#define C_   128
#define H_   1024
#define W_   16
#define W1_  17
#define SLOTS_ 80          // 16 seed frames + 64 decoder frames
#define EPS_ 1e-5f

typedef __bf16 bf16;
typedef __attribute__((ext_vector_type(16))) __bf16 v16bf;
typedef __attribute__((ext_vector_type(8)))  float  v8f;

#define WMMA_BF16(a, b, c) \
    __builtin_amdgcn_wmma_f32_16x16x32_bf16(false, (a), false, (b), (short)0, (c), false, false)

// =====================================================================
// Dual-problem WMMA GEMM:
//   D[M x N] = A[M x K](bf16, row-stride lda) @ Bw[N x K](bf16)^T + bias[N]
// blockIdx.z in {0,1} selects independent problem (gi / gh pair share one
// launch).  Block tile 64(M) x 128(N), 128 threads = 4 waves, each wave a
// 32x64 patch = 2x4 fragments = 8 v_wmma per K-chunk of 32.  LDS tiles are
// double-buffered (one barrier per chunk).  When mi_t >= 0 the A rows are
// gathered virtually from the sliding stream S (rows 0..15) and dec (row 16)
// — this folds the window-concat copy into the GEMM.
// =====================================================================
__global__ __launch_bounds__(128)
void wmma_gemm_k(const bf16* __restrict__ A0, int lda0, const bf16* __restrict__ Bw0,
                 const float* __restrict__ bias0, float* __restrict__ D0,
                 bf16* __restrict__ Dbf0, int K0,
                 const bf16* __restrict__ A1, int lda1, const bf16* __restrict__ Bw1,
                 const float* __restrict__ bias1, float* __restrict__ D1,
                 bf16* __restrict__ Dbf1, int K1,
                 int M, int N, const bf16* __restrict__ decb, int mi_t)
{
    __shared__ __attribute__((aligned(64))) bf16 lA[2][64][32];
    __shared__ __attribute__((aligned(64))) bf16 lB[2][128][32];

    const bf16* A;  int lda;  const bf16* Bw;  const float* bias;
    float* D;  bf16* Dbf;  int K;
    if (blockIdx.z == 0) { A = A0; lda = lda0; Bw = Bw0; bias = bias0; D = D0; Dbf = Dbf0; K = K0; }
    else                 { A = A1; lda = lda1; Bw = Bw1; bias = bias1; D = D1; Dbf = Dbf1; K = K1; }

    const int tid  = threadIdx.x;
    const int lane = tid & 31;
    const int wave = tid >> 5;
    const int bm = blockIdx.y * 64;
    const int bn = blockIdx.x * 128;
    const int wm = (wave & 1) * 32;        // wave row offset in block tile
    const int wn = (wave >> 1) * 64;       // wave col offset in block tile

    const int row_a = lane & 15;           // M (or N) index within 16
    const int koff  = (lane >> 4) * 16;    // K half selected by lane group

    v8f acc00 = {}, acc01 = {}, acc02 = {}, acc03 = {};
    v8f acc10 = {}, acc11 = {}, acc12 = {}, acc13 = {};

    // 16-byte vectorized tile staging (A: 256 uint4, B: 512 uint4)
    auto stage = [&](int buf, int k0) {
        for (int idx = tid; idx < 256; idx += 128) {
            int r = idx >> 2, c8 = (idx & 3) * 8;
            const bf16* arow;
            if (mi_t >= 0) {
                int gr = bm + r;
                int b = gr / W1_, j = gr - b * W1_;
                arow = (j < W_) ? A + ((size_t)b * SLOTS_ + mi_t + j) * C_
                                : decb + (size_t)b * C_;
            } else {
                arow = A + (size_t)(bm + r) * lda;
            }
            *(uint4*)&lA[buf][r][c8] = *(const uint4*)(arow + k0 + c8);
        }
        for (int idx = tid; idx < 512; idx += 128) {
            int r = idx >> 2, c8 = (idx & 3) * 8;
            *(uint4*)&lB[buf][r][c8] = *(const uint4*)(Bw + (size_t)(bn + r) * K + k0 + c8);
        }
    };

    const int nk = K >> 5;
    stage(0, 0);
    for (int kc = 0; kc < nk; ++kc) {
        __syncthreads();
        const int cur = kc & 1;

        v16bf a0 = *(const v16bf*)&lA[cur][wm      + row_a][koff];
        v16bf a1 = *(const v16bf*)&lA[cur][wm + 16 + row_a][koff];
        v16bf b0 = *(const v16bf*)&lB[cur][wn      + row_a][koff];
        v16bf b1 = *(const v16bf*)&lB[cur][wn + 16 + row_a][koff];
        v16bf b2 = *(const v16bf*)&lB[cur][wn + 32 + row_a][koff];
        v16bf b3 = *(const v16bf*)&lB[cur][wn + 48 + row_a][koff];

        if (kc + 1 < nk) stage(cur ^ 1, (kc + 1) * 32);

        acc00 = WMMA_BF16(a0, b0, acc00);
        acc01 = WMMA_BF16(a0, b1, acc01);
        acc02 = WMMA_BF16(a0, b2, acc02);
        acc03 = WMMA_BF16(a0, b3, acc03);
        acc10 = WMMA_BF16(a1, b0, acc10);
        acc11 = WMMA_BF16(a1, b1, acc11);
        acc12 = WMMA_BF16(a1, b2, acc12);
        acc13 = WMMA_BF16(a1, b3, acc13);
    }

    // C/D layout: VGPR j -> lanes 0-15: M=j, N=lane ; lanes 16-31: M=j+8, N=lane-16
    const int nlo = lane & 15;
    const int mhi = (lane >> 4) * 8;
    v8f* accs[2][4] = { { &acc00, &acc01, &acc02, &acc03 },
                        { &acc10, &acc11, &acc12, &acc13 } };
    for (int mt = 0; mt < 2; ++mt) {
        for (int nt = 0; nt < 4; ++nt) {
            const v8f& a = *accs[mt][nt];
            int c0 = bn + wn + nt * 16 + nlo;
            float bv = bias[c0];
            for (int j = 0; j < 8; ++j) {
                int r0 = bm + wm + mt * 16 + mhi + j;
                float v = a[j] + bv;
                D[(size_t)r0 * N + c0] = v;
                if (Dbf) Dbf[(size_t)r0 * N + c0] = (bf16)v;
            }
        }
    }
}

// ---------------- prep / pointwise kernels ----------------
__global__ void cvt_f32_bf16_k(const float* __restrict__ s, bf16* __restrict__ d, int n) {
    int i = blockIdx.x * blockDim.x + threadIdx.x;
    if (i < n) d[i] = (bf16)s[i];
}

__global__ void zero_f32_k(float* __restrict__ p, int n) {
    int i = blockIdx.x * blockDim.x + threadIdx.x;
    if (i < n) p[i] = 0.0f;
}

__global__ void zero_bf16_k(bf16* __restrict__ p, int n) {
    int i = blockIdx.x * blockDim.x + threadIdx.x;
    if (i < n) p[i] = (bf16)0.0f;
}

// seed stream S[0..15] = x[:, T-16:T, :]   (bf16 stream)
__global__ void init_S_k(const float* __restrict__ x, bf16* __restrict__ S) {
    int i = blockIdx.x * blockDim.x + threadIdx.x;     // B*16*C
    if (i >= B_ * W_ * C_) return;
    int c = i % C_, j = (i / C_) % W_, b = i / (C_ * W_);
    S[((size_t)b * SLOTS_ + j) * C_ + c] = (bf16)x[((size_t)b * T_ + (T_ - W_ + j)) * C_ + c];
}

__global__ void init_decin_k(const float* __restrict__ x, float* __restrict__ dec_in,
                             bf16* __restrict__ decin_b) {
    int i = blockIdx.x * blockDim.x + threadIdx.x;     // B*C
    if (i >= B_ * C_) return;
    int c = i % C_, b = i / C_;
    float v = x[((size_t)b * T_ + (T_ - 1)) * C_ + c];
    dec_in[i] = v;
    decin_b[i] = (bf16)v;
}

// GRU gate fusion: h = (1-z)*n + z*h ; also writes bf16 mirror for next GEMM
__global__ void gru_gate_k(const float* __restrict__ gi, const float* __restrict__ gh,
                           float* __restrict__ h, bf16* __restrict__ hb) {
    int i = blockIdx.x * blockDim.x + threadIdx.x;     // B*H
    if (i >= B_ * H_) return;
    int j = i % H_, b = i / H_;
    const float* gib = gi + (size_t)b * 3 * H_;
    const float* ghb = gh + (size_t)b * 3 * H_;
    float r  = 1.0f / (1.0f + __expf(-(gib[j] + ghb[j])));
    float z  = 1.0f / (1.0f + __expf(-(gib[H_ + j] + ghb[H_ + j])));
    float n  = tanhf(gib[2 * H_ + j] + r * ghb[2 * H_ + j]);
    float hn = (1.0f - z) * n + z * h[i];
    h[i]  = hn;
    hb[i] = (bf16)hn;
}

// x_[b, c, wo] = sum_w f[b, w, c] * Wmlp[wo, w] + bmlp[wo]
__global__ __launch_bounds__(128)
void mlp_w1_k(const float* __restrict__ f, const float* __restrict__ Wmlp,
              const float* __restrict__ bmlp, float* __restrict__ xo) {
    __shared__ float sW[W1_ * W1_];
    __shared__ float sB[W1_];
    for (int idx = threadIdx.x; idx < W1_ * W1_; idx += 128) sW[idx] = Wmlp[idx];
    for (int idx = threadIdx.x; idx < W1_; idx += 128) sB[idx] = bmlp[idx];
    __syncthreads();
    int i = blockIdx.x * blockDim.x + threadIdx.x;     // B*C
    if (i >= B_ * C_) return;
    int c = i % C_, b = i / C_;
    float tv[W1_];
    for (int w = 0; w < W1_; ++w) tv[w] = f[((size_t)b * W1_ + w) * C_ + c];
    float* op = xo + ((size_t)b * C_ + c) * W1_;
    for (int wo = 0; wo < W1_; ++wo) {
        float s = sB[wo];
        const float* wr = &sW[wo * W1_];
        for (int w = 0; w < W1_; ++w) s += tv[w] * wr[w];
        op[wo] = s;
    }
}

// layernorm over C axis of x_[b, c, wo]; g2b[b, wo, c] = bf16(f[b, wo, c] + y)
__global__ void ln_k(const float* __restrict__ xo, const float* __restrict__ f,
                     const float* __restrict__ alpha, const float* __restrict__ beta,
                     bf16* __restrict__ g2b) {
    int i = blockIdx.x * blockDim.x + threadIdx.x;     // B*W1
    if (i >= B_ * W1_) return;
    int wo = i % W1_, b = i / W1_;
    const float* xb = xo + (size_t)b * C_ * W1_ + wo;
    float sum = 0.f, sumsq = 0.f;
    for (int c = 0; c < C_; ++c) { float v = xb[(size_t)c * W1_]; sum += v; sumsq += v * v; }
    float mean = sum * (1.0f / C_);
    float var  = sumsq * (1.0f / C_) - mean * mean;
    float rstd = rsqrtf(var + EPS_);
    const float* fb = f + ((size_t)b * W1_ + wo) * C_;
    bf16* gb = g2b + ((size_t)b * W1_ + wo) * C_;
    for (int c = 0; c < C_; ++c) {
        float v = xb[(size_t)c * W1_];
        float y = (v - mean) * rstd * alpha[c] + beta[c];
        gb[c] = (bf16)(fb[c] + y);
    }
}

// nf = f2 . Wm + bm + dec_in ; update dec_in (f32 + bf16), stream S, output
__global__ void merge_k(const float* __restrict__ f2, const float* __restrict__ Wm,
                        const float* __restrict__ bm, float* __restrict__ dec_in,
                        bf16* __restrict__ decin_b, bf16* __restrict__ S,
                        float* __restrict__ out, int t) {
    int i = blockIdx.x * blockDim.x + threadIdx.x;     // B*C
    if (i >= B_ * C_) return;
    int c = i % C_, b = i / C_;
    float s = bm[0];
    for (int w = 0; w < W1_; ++w) s += f2[((size_t)b * W1_ + w) * C_ + c] * Wm[w];
    float nf = s + dec_in[i];
    dec_in[i]  = nf;
    decin_b[i] = (bf16)nf;
    S[((size_t)b * SLOTS_ + W_ + t) * C_ + c] = (bf16)nf;
    out[((size_t)b * T_ + t) * C_ + c] = nf;
}

// =====================================================================
// host orchestration
// =====================================================================
static inline char* ws_take(char*& p, size_t bytes) {
    char* r = p;
    p += (bytes + 255) & ~(size_t)255;
    return r;
}

extern "C" void kernel_launch(void* const* d_in, const int* in_sizes, int n_in,
                              void* d_out, int out_size, void* d_ws, size_t ws_size,
                              hipStream_t stream) {
    (void)in_sizes; (void)n_in; (void)out_size; (void)ws_size;

    const float* x    = (const float*)d_in[0];
    const float* Wih0 = (const float*)d_in[1];
    const float* Whh0 = (const float*)d_in[2];
    const float* bih0 = (const float*)d_in[3];
    const float* bhh0 = (const float*)d_in[4];
    const float* Wih1 = (const float*)d_in[5];
    const float* Whh1 = (const float*)d_in[6];
    const float* bih1 = (const float*)d_in[7];
    const float* bhh1 = (const float*)d_in[8];
    const float* Wdec = (const float*)d_in[9];
    const float* bdec = (const float*)d_in[10];
    const float* Wfi  = (const float*)d_in[11];
    const float* bfi  = (const float*)d_in[12];
    const float* Wmlp = (const float*)d_in[13];
    const float* bmlp = (const float*)d_in[14];
    const float* alpha= (const float*)d_in[15];
    const float* beta = (const float*)d_in[16];
    const float* Wfo  = (const float*)d_in[17];
    const float* bfo  = (const float*)d_in[18];
    const float* Wm   = (const float*)d_in[19];
    const float* bm   = (const float*)d_in[20];
    float* out = (float*)d_out;

    // ---- workspace carve-up ----
    char* p = (char*)d_ws;
    bf16* wWih0 = (bf16*)ws_take(p, (size_t)3 * H_ * C_ * sizeof(bf16));
    bf16* wWhh0 = (bf16*)ws_take(p, (size_t)3 * H_ * H_ * sizeof(bf16));
    bf16* wWih1 = (bf16*)ws_take(p, (size_t)3 * H_ * H_ * sizeof(bf16));
    bf16* wWhh1 = (bf16*)ws_take(p, (size_t)3 * H_ * H_ * sizeof(bf16));
    bf16* wWdec = (bf16*)ws_take(p, (size_t)C_ * H_ * sizeof(bf16));
    bf16* wWfi  = (bf16*)ws_take(p, (size_t)C_ * C_ * sizeof(bf16));
    bf16* wWfo  = (bf16*)ws_take(p, (size_t)C_ * C_ * sizeof(bf16));
    bf16* xb     = (bf16*)ws_take(p, (size_t)B_ * T_ * C_ * sizeof(bf16));
    float* h0    = (float*)ws_take(p, (size_t)B_ * H_ * sizeof(float));
    float* h1    = (float*)ws_take(p, (size_t)B_ * H_ * sizeof(float));
    bf16* hb0    = (bf16*)ws_take(p, (size_t)B_ * H_ * sizeof(bf16));
    bf16* hb1    = (bf16*)ws_take(p, (size_t)B_ * H_ * sizeof(bf16));
    float* gi    = (float*)ws_take(p, (size_t)B_ * 3 * H_ * sizeof(float));
    float* gh    = (float*)ws_take(p, (size_t)B_ * 3 * H_ * sizeof(float));
    bf16* S      = (bf16*)ws_take(p, (size_t)B_ * SLOTS_ * C_ * sizeof(bf16));
    float* dec   = (float*)ws_take(p, (size_t)B_ * C_ * sizeof(float));
    bf16* decb   = (bf16*)ws_take(p, (size_t)B_ * C_ * sizeof(bf16));
    float* dec_in= (float*)ws_take(p, (size_t)B_ * C_ * sizeof(float));
    bf16* decin_b= (bf16*)ws_take(p, (size_t)B_ * C_ * sizeof(bf16));
    float* fbuf  = (float*)ws_take(p, (size_t)B_ * W1_ * C_ * sizeof(float));
    float* xobuf = (float*)ws_take(p, (size_t)B_ * C_ * W1_ * sizeof(float));
    bf16* g2b    = (bf16*)ws_take(p, (size_t)B_ * W1_ * C_ * sizeof(bf16));
    float* f2    = (float*)ws_take(p, (size_t)B_ * W1_ * C_ * sizeof(float));

    auto cvt = [&](const float* s, bf16* d, int n) {
        cvt_f32_bf16_k<<<(n + 255) / 256, 256, 0, stream>>>(s, d, n);
    };
    // dual independent GEMMs in one launch (z = 0: problem a, z = 1: problem b)
    auto gemm2 = [&](const bf16* Aa, int ldaa, const bf16* Ba, const float* biasa, float* Da, int Ka,
                     const bf16* Ab, int ldab, const bf16* Bb, const float* biasb, float* Db, int Kb,
                     int M, int N) {
        dim3 grid(N / 128, M / 64, 2);
        wmma_gemm_k<<<grid, 128, 0, stream>>>(Aa, ldaa, Ba, biasa, Da, (bf16*)nullptr, Ka,
                                              Ab, ldab, Bb, biasb, Db, (bf16*)nullptr, Kb,
                                              M, N, (const bf16*)nullptr, -1);
    };
    auto gemm1 = [&](const bf16* A, int lda, const bf16* Bw, const float* bias,
                     float* D, bf16* Dbf, int M, int N, int K,
                     const bf16* decp, int mi_t) {
        dim3 grid(N / 128, M / 64, 1);
        wmma_gemm_k<<<grid, 128, 0, stream>>>(A, lda, Bw, bias, D, Dbf, K,
                                              A, lda, Bw, bias, D, Dbf, K,
                                              M, N, decp, mi_t);
    };

    // ---- one-time prep (bf16 weights + activations; weights stay in L2) ----
    cvt(Wih0, wWih0, 3 * H_ * C_);
    cvt(Whh0, wWhh0, 3 * H_ * H_);
    cvt(Wih1, wWih1, 3 * H_ * H_);
    cvt(Whh1, wWhh1, 3 * H_ * H_);
    cvt(Wdec, wWdec, C_ * H_);
    cvt(Wfi,  wWfi,  C_ * C_);
    cvt(Wfo,  wWfo,  C_ * C_);
    cvt(x,    xb,    B_ * T_ * C_);
    zero_f32_k <<<(B_ * H_ + 255) / 256, 256, 0, stream>>>(h0,  B_ * H_);
    zero_f32_k <<<(B_ * H_ + 255) / 256, 256, 0, stream>>>(h1,  B_ * H_);
    zero_bf16_k<<<(B_ * H_ + 255) / 256, 256, 0, stream>>>(hb0, B_ * H_);
    zero_bf16_k<<<(B_ * H_ + 255) / 256, 256, 0, stream>>>(hb1, B_ * H_);
    init_S_k<<<(B_ * W_ * C_ + 255) / 256, 256, 0, stream>>>(x, S);
    init_decin_k<<<(B_ * C_ + 255) / 256, 256, 0, stream>>>(x, dec_in, decin_b);

    const int gateBlocks = (B_ * H_ + 255) / 256;

    // ---- encoder scan ----
    for (int t = 0; t < T_; ++t) {
        gemm2(xb + (size_t)t * C_, T_ * C_, wWih0, bih0, gi, C_,
              hb0, H_,                      wWhh0, bhh0, gh, H_, B_, 3 * H_);
        gru_gate_k<<<gateBlocks, 256, 0, stream>>>(gi, gh, h0, hb0);
        gemm2(hb0, H_, wWih1, bih1, gi, H_,
              hb1, H_, wWhh1, bhh1, gh, H_, B_, 3 * H_);
        gru_gate_k<<<gateBlocks, 256, 0, stream>>>(gi, gh, h1, hb1);
    }

    // ---- decoder scan ----
    for (int t = 0; t < T_; ++t) {
        gemm2(decin_b, C_, wWih0, bih0, gi, C_,
              hb0, H_,     wWhh0, bhh0, gh, H_, B_, 3 * H_);
        gru_gate_k<<<gateBlocks, 256, 0, stream>>>(gi, gh, h0, hb0);
        gemm2(hb0, H_, wWih1, bih1, gi, H_,
              hb1, H_, wWhh1, bhh1, gh, H_, B_, 3 * H_);
        gru_gate_k<<<gateBlocks, 256, 0, stream>>>(gi, gh, h1, hb1);

        // dec = h1 @ Wdec.T + bdec  (bf16 mirror feeds the window GEMM)
        gemm1(hb1, H_, wWdec, bdec, dec, decb, B_, C_, H_, nullptr, -1);

        // f = [S[t:t+16] | dec] @ Wfi.T + bfi   (window gather fused into GEMM)
        gemm1(S, C_, wWfi, bfi, fbuf, nullptr, B_ * W1_, C_, C_, decb, t);
        mlp_w1_k<<<(B_ * C_ + 127) / 128, 128, 0, stream>>>(fbuf, Wmlp, bmlp, xobuf);
        ln_k<<<(B_ * W1_ + 255) / 256, 256, 0, stream>>>(xobuf, fbuf, alpha, beta, g2b);
        gemm1(g2b, C_, wWfo, bfo, f2, nullptr, B_ * W1_, C_, C_, nullptr, -1);
        merge_k<<<(B_ * C_ + 255) / 256, 256, 0, stream>>>(f2, Wm, bm, dec_in, decin_b, S, out, t);
    }
}
